// TritonGatherConv_82429012344832
// MI455X (gfx1250) — matrix operand order repaired and checked by reading, not compiled, not run on hardware
//
#include <hip/hip_runtime.h>
#include <hip/hip_bf16.h>
#include <math.h>

typedef __bf16 bf16;
typedef __attribute__((ext_vector_type(16))) __bf16 v16bf;
typedef __attribute__((ext_vector_type(8)))  __bf16 v8bf;
typedef __attribute__((ext_vector_type(4)))  __bf16 v4bf;
typedef __attribute__((ext_vector_type(8)))  float  v8f;
typedef __attribute__((ext_vector_type(4)))  unsigned int v4u;
typedef __attribute__((ext_vector_type(8)))  int    v8i;
typedef __attribute__((ext_vector_type(4)))  int    v4i;

#define B_      2
#define L_      2048
#define C_      1024
#define H_      8
#define D_      128
#define S_      33
#define HALFS_  16
#define NW_     16          // 2*H
#define NK_     512         // H*K
#define M_      (B_*L_)     // 4096

__device__ __forceinline__ float sigmoidf_(float v) { return 1.f / (1.f + __expf(-v)); }
__device__ __forceinline__ float siluf_(float v)    { return v * sigmoidf_(v); }

// ---- WMMA fragment loaders (bf16, 16x16x32, wave32) -------------------------
// A (16x32, MxK) from row-major [M,K]: lane(0..15)=row M, half selects K-phase.
__device__ __forceinline__ v16bf load_a_frag(const bf16* A, int lda, int m0, int k0) {
    const int lane = threadIdx.x & 31;
    const int half = lane >> 4;
    const int m    = lane & 15;
    const bf16* p  = A + (size_t)(m0 + m) * lda + k0 + half * 16;
    v8bf lo = *(const v8bf*)(p);
    v8bf hi = *(const v8bf*)(p + 8);
    v16bf r;
#pragma unroll
    for (int i = 0; i < 8; ++i) { r[i] = lo[i]; r[8 + i] = hi[i]; }
    return r;
}

// B fragment (32x16) from a per-lane row pointer into W[N][K] (Bmat[k][n]=W[n][k]):
// lane's row = W + (n0 + lane&15)*ldw, k chunk at k0 + half*16 (32 contiguous bytes).
__device__ __forceinline__ v16bf load_b_frag_row(const bf16* row, int k0) {
    const int lane = threadIdx.x & 31;
    const int half = lane >> 4;
    const bf16* p  = row + k0 + half * 16;
    v8bf lo = *(const v8bf*)(p);
    v8bf hi = *(const v8bf*)(p + 8);
    v16bf r;
#pragma unroll
    for (int i = 0; i < 8; ++i) { r[i] = lo[i]; r[8 + i] = hi[i]; }
    return r;
}

__device__ __forceinline__ v8f wmma_bf16(v16bf a, v16bf b, v8f c) {
    return __builtin_amdgcn_wmma_f32_16x16x32_bf16(
        false, a, false, b, (short)0, c, false, false);
}

// ---- TDM staging: DMA a contiguous `bytes` block (global -> LDS) ------------
// One wave issues a 1-D tensor descriptor (data_size = 4B dwords); the rest of
// the workgroup is released by the barrier after s_wait_tensorcnt 0.
// D# group0: [1:0]=count=1, [63:32]=lds_addr, [120:64]=global_addr, [127:126]=type=2.
// D# group1: [17:16]=data_size=2(4B), [63:48]=tensor_dim0.lo, [79:64]=tensor_dim0.hi,
//            [95:80]=tensor_dim1=1, [127:112]=tile_dim0, [191:160]=tensor_dim0_stride.
__device__ __forceinline__ void stage_tile_lds(const bf16* gsrc, bf16* lds_dst, int bytes) {
#if __has_builtin(__builtin_amdgcn_tensor_load_to_lds) && __has_builtin(__builtin_amdgcn_s_wait_tensorcnt)
    if (threadIdx.x < 32) {
        // flat->LDS mapping: LDS address is the low 32 bits of the flat address (ISA 10.2), so low 32 bits
        // of the generic __shared__ pointer are the descriptor's lds_addr.
        unsigned lds_addr = (unsigned)(size_t)lds_dst;
        unsigned long long ga = (unsigned long long)(size_t)gsrc;
        const unsigned ndw = (unsigned)(bytes >> 2);          // dwords
        v4u g0;
        g0[0] = 1u;                                           // count=1
        g0[1] = lds_addr;                                     // lds_addr[31:0]
        g0[2] = (unsigned)(ga & 0xFFFFFFFFu);                 // global_addr[31:0]
        g0[3] = (unsigned)((ga >> 32) & 0x1FFFFFFu) | (2u << 30); // addr[56:32] | type=2
        v8i g1;
        g1[0] = (int)(2u << 16);                              // data_size = 4B
        g1[1] = (int)((ndw & 0xFFFFu) << 16);                 // tensor_dim0[15:0]
        g1[2] = (int)((ndw >> 16) | (1u << 16));              // tensor_dim0[31:16] | tensor_dim1=1
        g1[3] = (int)((ndw & 0xFFFFu) << 16);                 // tile_dim0 (1-D tile)
        g1[4] = 0;                                            // tile_dim1/2 unused
        g1[5] = (int)ndw;                                     // tensor_dim0_stride[31:0]
        g1[6] = 0;
        g1[7] = 0;
        v4i z4 = {0, 0, 0, 0};
        v8i z8 = {0, 0, 0, 0, 0, 0, 0, 0};
        // 6-arg form (clang-23 / therock headers): groups 0..3 + extra group + cpol
        __builtin_amdgcn_tensor_load_to_lds(g0, g1, z4, z4, z8, 0);
        __builtin_amdgcn_s_wait_tensorcnt((short)0);
    }
    __syncthreads();
#else
    const v8bf* src = (const v8bf*)gsrc;
    v8bf* dst = (v8bf*)lds_dst;
    for (int i = threadIdx.x; i < bytes / 16; i += (int)blockDim.x) dst[i] = src[i];
    __syncthreads();
#endif
}

// ---- f32 -> bf16 conversion -------------------------------------------------
__global__ void cvt_f32_bf16(const float* __restrict__ s, bf16* __restrict__ d, int n) {
    int i = blockIdx.x * 256 + threadIdx.x;
    if (i < n) d[i] = (bf16)s[i];
}

// ---- GEMM 1: wave projection (N=16) -> freq/phase ---------------------------
__global__ void wave_gemm(const bf16* __restrict__ Xbf, const bf16* __restrict__ WWbf,
                          const float* __restrict__ wb,
                          float* __restrict__ freq, float* __restrict__ phase) {
    const int m0 = blockIdx.x * 16;
    const bf16* brow = WWbf + (size_t)(threadIdx.x & 15) * C_;
    v8f acc = {};
#pragma unroll
    for (int k = 0; k < C_; k += 32) {
        v16bf a = load_a_frag(Xbf, C_, m0, k);
        v16bf b = load_b_frag_row(brow, k);
        acc = wmma_bf16(a, b, acc);
    }
    const int lane = threadIdx.x & 31, half = lane >> 4, n = lane & 15;
    const float bias = wb[n];
#pragma unroll
    for (int r = 0; r < 8; ++r) {
        int m = m0 + half * 8 + r;
        float v = siluf_(acc[r] + bias);
        if (n < H_) freq[m * H_ + n]         = sigmoidf_(v) * 15.f + 1.f;
        else        phase[m * H_ + (n - H_)] = tanhf(v) * 16.f;
    }
}

// ---- GEMM 2: tap-kernel projection (N=512) -> kern[B*L,H,S] -----------------
// 8 waves/block, each wave owns a 16x64 strip (4 accumulators, A reused 4x).
__global__ __launch_bounds__(256) void kern_gemm(const bf16* __restrict__ Xbf,
                                                 const bf16* __restrict__ KWbf,
                                                 const float* __restrict__ kb,
                                                 float* __restrict__ kern) {
    __shared__ bf16 As[16 * C_];                       // 32 KB
    const int m0 = blockIdx.x * 16;
    stage_tile_lds(Xbf + (size_t)m0 * C_, As, 16 * C_ * (int)sizeof(bf16));

    const int wave = threadIdx.x >> 5;
    const int n0   = wave * 64;                        // 8 waves * 64 = 512 cols
    const bf16* brow0 = KWbf + (size_t)(n0 + (threadIdx.x & 15)) * C_;
    v8f acc[4] = {};
#pragma unroll
    for (int k = 0; k < C_; k += 32) {
        v16bf a = load_a_frag(As, C_, 0, k);           // ds_load from LDS
#pragma unroll
        for (int t = 0; t < 4; ++t) {
            v16bf b = load_b_frag_row(brow0 + (size_t)t * 16 * C_, k);
            acc[t] = wmma_bf16(a, b, acc[t]);
        }
    }
    const int lane = threadIdx.x & 31, half = lane >> 4, n = lane & 15;
#pragma unroll
    for (int t = 0; t < 4; ++t) {
        const int col = n0 + t * 16 + n, h = col >> 6, tap = col & 63;
        if (tap < S_) {
            const float bias = kb[col];
#pragma unroll
            for (int r = 0; r < 8; ++r) {
                int m = m0 + half * 8 + r;
                kern[((size_t)m * H_ + h) * S_ + tap] = siluf_(acc[t][r] + bias);
            }
        }
    }
}

// ---- gather / lerp / tap accumulation: one wave per (b,h,l) -----------------
__global__ __launch_bounds__(256) void gather_kernel(const float* __restrict__ x,
                                                     const float* __restrict__ freq,
                                                     const float* __restrict__ phase,
                                                     const float* __restrict__ kern,
                                                     bf16* __restrict__ hidden) {
    const int wave = threadIdx.x >> 5;
    const int lane = threadIdx.x & 31;
    const int g  = blockIdx.x * 8 + wave;      // ((b*H)+h)*L + l
    const int l  = g & (L_ - 1);
    const int bh = g >> 11;
    const int h  = bh & (H_ - 1);
    const int b  = bh >> 3;
    const int m  = b * L_ + l;

    const float fv = freq[m * H_ + h];
    const float pv = phase[m * H_ + h];
    const float* kr = kern + ((size_t)m * H_ + h) * S_;
    const float* xbase = x + (size_t)b * L_ * C_ + h * D_;

    float a0 = 0.f, a1 = 0.f, a2 = 0.f, a3 = 0.f;
    for (int s = 0; s < S_; ++s) {
        float pos = (float)l + pv + (float)(s - HALFS_) * fv;
        pos = fminf(fmaxf(pos, 0.f), (float)(L_ - 1));
        int p0 = (int)pos;                     // pos >= 0 -> trunc == floor
        int p1 = p0 + 1; if (p1 > L_ - 1) p1 = L_ - 1;
        float w1 = pos - (float)p0;
        float w0 = 1.f - w1;
        float kv = kr[s];
        float4 r0 = ((const float4*)(xbase + (size_t)p0 * C_))[lane];
        float4 r1 = ((const float4*)(xbase + (size_t)p1 * C_))[lane];
        a0 += kv * (w0 * r0.x + w1 * r1.x);
        a1 += kv * (w0 * r0.y + w1 * r1.y);
        a2 += kv * (w0 * r0.z + w1 * r1.z);
        a3 += kv * (w0 * r0.w + w1 * r1.w);
    }
    v4bf o; o[0] = (bf16)a0; o[1] = (bf16)a1; o[2] = (bf16)a2; o[3] = (bf16)a3;
    *(v4bf*)(hidden + (size_t)m * C_ + h * D_ + lane * 4) = o;
}

// ---- GEMM 3: out projection (N=1024) + SiLU -> d_out (f32) ------------------
// 8 waves/block, each wave owns a 16x64 strip (4 accumulators, A reused 4x).
__global__ __launch_bounds__(256) void out_gemm(const bf16* __restrict__ Abf,
                                                const bf16* __restrict__ OWbf,
                                                float* __restrict__ out) {
    __shared__ bf16 As[16 * C_];                       // 32 KB
    const int m0 = blockIdx.x * 16;
    stage_tile_lds(Abf + (size_t)m0 * C_, As, 16 * C_ * (int)sizeof(bf16));

    const int wave = threadIdx.x >> 5;
    const int n0   = (blockIdx.y * 8 + wave) * 64;     // 2 * 8 waves * 64 = 1024 cols
    const bf16* brow0 = OWbf + (size_t)(n0 + (threadIdx.x & 15)) * C_;
    v8f acc[4] = {};
#pragma unroll
    for (int k = 0; k < C_; k += 32) {
        v16bf a = load_a_frag(As, C_, 0, k);
#pragma unroll
        for (int t = 0; t < 4; ++t) {
            v16bf b = load_b_frag_row(brow0 + (size_t)t * 16 * C_, k);
            acc[t] = wmma_bf16(a, b, acc[t]);
        }
    }
    const int lane = threadIdx.x & 31, half = lane >> 4, n = lane & 15;
#pragma unroll
    for (int t = 0; t < 4; ++t)
#pragma unroll
        for (int r = 0; r < 8; ++r) {
            int m = m0 + half * 8 + r;
            out[(size_t)m * C_ + n0 + t * 16 + n] = siluf_(acc[t][r]);
        }
}

// -----------------------------------------------------------------------------
extern "C" void kernel_launch(void* const* d_in, const int* in_sizes, int n_in,
                              void* d_out, int out_size, void* d_ws, size_t ws_size,
                              hipStream_t stream) {
    (void)in_sizes; (void)n_in; (void)out_size; (void)ws_size;

    const float* x        = (const float*)d_in[0];   // [B,L,C]
    const float* wave_w   = (const float*)d_in[1];   // [2H,C]
    const float* wave_b   = (const float*)d_in[2];   // [2H]
    const float* kernel_w = (const float*)d_in[3];   // [H*K,C]
    const float* kernel_b = (const float*)d_in[4];   // [H*K]
    const float* out_w    = (const float*)d_in[5];   // [C,C]
    float* out = (float*)d_out;                      // [B,L,C] f32

    char* ws = (char*)d_ws;
    size_t off = 0;
    auto take = [&](size_t bytes) -> char* {
        char* p = ws + off;
        off += (bytes + 255) & ~(size_t)255;
        return p;
    };
    bf16*  x_bf   = (bf16*) take((size_t)M_ * C_  * sizeof(bf16));   // 8 MiB
    bf16*  ww_bf  = (bf16*) take((size_t)NW_ * C_ * sizeof(bf16));
    bf16*  kw_bf  = (bf16*) take((size_t)NK_ * C_ * sizeof(bf16));
    bf16*  ow_bf  = (bf16*) take((size_t)C_ * C_  * sizeof(bf16));
    float* freq   = (float*)take((size_t)M_ * H_  * sizeof(float));
    float* phase  = (float*)take((size_t)M_ * H_  * sizeof(float));
    float* kern   = (float*)take((size_t)M_ * H_ * S_ * sizeof(float));
    bf16*  hid_bf = (bf16*) take((size_t)M_ * C_  * sizeof(bf16));   // 8 MiB

    // 1) bf16 operand conversion for the WMMA GEMMs
    cvt_f32_bf16<<<(M_ * C_  + 255) / 256, 256, 0, stream>>>(x,        x_bf,  M_ * C_);
    cvt_f32_bf16<<<(NW_ * C_ + 255) / 256, 256, 0, stream>>>(wave_w,   ww_bf, NW_ * C_);
    cvt_f32_bf16<<<(NK_ * C_ + 255) / 256, 256, 0, stream>>>(kernel_w, kw_bf, NK_ * C_);
    cvt_f32_bf16<<<(C_ * C_  + 255) / 256, 256, 0, stream>>>(out_w,    ow_bf, C_ * C_);

    // 2) wave projection -> freq/phase   (256 single-wave tiles)
    wave_gemm<<<M_ / 16, 32, 0, stream>>>(x_bf, ww_bf, wave_b, freq, phase);

    // 3) tap-kernel projection -> kern   (256 blocks, 8 waves x 64 cols)
    kern_gemm<<<dim3(M_ / 16, 1), 256, 0, stream>>>(x_bf, kw_bf, kernel_b, kern);

    // 4) gather + lerp + tap accumulation -> hidden (bf16)
    gather_kernel<<<(B_ * H_ * L_) / 8, 256, 0, stream>>>(x, freq, phase, kern, hid_bf);

    // 5) out projection + SiLU -> d_out  (256 x 2 blocks, 8 waves x 64 cols)
    out_gemm<<<dim3(M_ / 16, C_ / 512), 256, 0, stream>>>(hid_bf, ow_bf, out);
}
